// AttentionLayerWithCoverage_25933012533308
// MI455X (gfx1250) — matrix operand order repaired and checked by reading, not compiled, not run on hardware
//
#include <hip/hip_runtime.h>
#include <math.h>

typedef __bf16 bf16;
typedef __attribute__((ext_vector_type(16))) __bf16 v16bf;
typedef __attribute__((ext_vector_type(8)))  __bf16 v8bf;
typedef __attribute__((ext_vector_type(8)))  float  v8f;

#define B_  32
#define T_  4096
#define DE_ 256
#define DD_ 128
#define LDSTRIDE 136   // padded bf16 row stride: 272B rows -> 16B aligned, 4-bank skew

// ---------------------------------------------------------------------------
// Stage a 128x128 K-block of a [K x 128] row-major fp32 weight matrix into LDS,
// transposed to [n][k] bf16 with padded stride. Bt[n*136 + k] = W[(k+k0)*128+n].
// ---------------------------------------------------------------------------
__device__ __forceinline__ void stage_weights(const float* __restrict__ Wsrc, int k0,
                                              bf16* Bt, int tid)
{
    for (int idx = tid; idx < 128 * 128; idx += 256) {
        int k = idx >> 7;
        int n = idx & 127;
        Bt[n * LDSTRIDE + k] = (bf16)Wsrc[(size_t)(k + k0) * 128 + n];
    }
}

// A fragment: 16x32 bf16 from an fp32 global row.
// Layout: lane<16 -> K base 0 (runs 0-7 & 16-23), lane>=16 -> base 8 (8-15 & 24-31).
__device__ __forceinline__ v16bf load_afrag(const float* __restrict__ arow, int kA, int half)
{
    v16bf f;
    const int base = kA + half * 8;
#pragma unroll
    for (int p = 0; p < 8; ++p) {
        f[p]     = (bf16)arow[base + p];
        f[p + 8] = (bf16)arow[base + 16 + p];
    }
    return f;
}

// B fragment: 32x16 bf16 from LDS (lanes 0-15: K 0-7 & 16-23, lanes 16-31: K 8-15 & 24-31).
__device__ __forceinline__ v16bf load_bfrag(const bf16* Bt, int j, int kB, int lrow, int half)
{
    const bf16* wp = Bt + (j * 16 + lrow) * LDSTRIDE + kB + half * 8;
    v8bf lo = *(const v8bf*)wp;
    v8bf hi = *(const v8bf*)(wp + 16);
    v16bf f;
#pragma unroll
    for (int p = 0; p < 8; ++p) { f[p] = lo[p]; f[p + 8] = hi[p]; }
    return f;
}

#define WMMA_BF16(A, Bf, C) \
    __builtin_amdgcn_wmma_f32_16x16x32_bf16(false, (A), false, (Bf), (short)0, (C), false, false)

// One K=32 step for TWO M-tiles sharing each B fragment (2x reuse).
// B fragments are loaded in pairs so ds_loads clause and WMMAs go back-to-back.
__device__ __forceinline__ void wmma_step2(const float* __restrict__ arow0,
                                           const float* __restrict__ arow1, int kA,
                                           const bf16* Bt, int kB,
                                           int lrow, int half, v8f acc0[8], v8f acc1[8])
{
    v16bf a0 = load_afrag(arow0, kA, half);
    v16bf a1 = load_afrag(arow1, kA, half);
#pragma unroll
    for (int j = 0; j < 8; j += 2) {
        v16bf b0 = load_bfrag(Bt, j,     kB, lrow, half);
        v16bf b1 = load_bfrag(Bt, j + 1, kB, lrow, half);
        acc0[j]     = WMMA_BF16(a0, b0, acc0[j]);
        acc1[j]     = WMMA_BF16(a1, b0, acc1[j]);
        acc0[j + 1] = WMMA_BF16(a0, b1, acc0[j + 1]);
        acc1[j + 1] = WMMA_BF16(a1, b1, acc1[j + 1]);
    }
}

// ---------------------------------------------------------------------------
// Pass 1: e[b,t] = v_a . tanh(enc[b,t,:]@W_a + dec[b,t,:]@U_a)
// Grid: (T/256, B), block 256 (8 waves), each wave owns 32 timesteps (2 M-tiles).
// ---------------------------------------------------------------------------
__global__ __launch_bounds__(256)
void score_kernel(const float* __restrict__ enc, const float* __restrict__ dec,
                  const float* __restrict__ W_a, const float* __restrict__ U_a,
                  const float* __restrict__ v_a, float* __restrict__ e_out)
{
    __shared__ __align__(16) bf16 Bt[128 * LDSTRIDE];

    const int tid    = threadIdx.x;
    const int b      = blockIdx.y;
    const int tblock = blockIdx.x * 256;
    const int wave   = tid >> 5;
    const int lane   = tid & 31;
    const int lrow   = lane & 15;
    const int half   = lane >> 4;
    const int twave  = tblock + wave * 32;

    const float* arow_e0 = enc + ((size_t)b * T_ + twave + lrow) * DE_;
    const float* arow_e1 = enc + ((size_t)b * T_ + twave + 16 + lrow) * DE_;
    const float* arow_d0 = dec + ((size_t)b * T_ + twave + lrow) * DD_;
    const float* arow_d1 = dec + ((size_t)b * T_ + twave + 16 + lrow) * DD_;

    v8f acc0[8], acc1[8];
#pragma unroll
    for (int j = 0; j < 8; ++j)
#pragma unroll
        for (int r = 0; r < 8; ++r) { acc0[j][r] = 0.0f; acc1[j][r] = 0.0f; }

    // enc @ W_a : K = 0..127
    stage_weights(W_a, 0, Bt, tid);
    __syncthreads();
#pragma unroll
    for (int kc = 0; kc < 128; kc += 32)
        wmma_step2(arow_e0, arow_e1, kc, Bt, kc, lrow, half, acc0, acc1);
    __syncthreads();

    // enc @ W_a : K = 128..255
    stage_weights(W_a, 128, Bt, tid);
    __syncthreads();
#pragma unroll
    for (int kc = 0; kc < 128; kc += 32)
        wmma_step2(arow_e0, arow_e1, 128 + kc, Bt, kc, lrow, half, acc0, acc1);
    __syncthreads();

    // dec @ U_a : K = 0..127
    stage_weights(U_a, 0, Bt, tid);
    __syncthreads();
#pragma unroll
    for (int kc = 0; kc < 128; kc += 32)
        wmma_step2(arow_d0, arow_d1, kc, Bt, kc, lrow, half, acc0, acc1);

    // Epilogue: tanh + dot with v_a, reduce across the 16 lanes of each half.
    float va[8];
#pragma unroll
    for (int j = 0; j < 8; ++j) va[j] = v_a[j * 16 + lrow];

    float p0[8], p1[8];
#pragma unroll
    for (int r = 0; r < 8; ++r) { p0[r] = 0.0f; p1[r] = 0.0f; }
#pragma unroll
    for (int j = 0; j < 8; ++j)
#pragma unroll
        for (int r = 0; r < 8; ++r) {
            p0[r] += tanhf(acc0[j][r]) * va[j];
            p1[r] += tanhf(acc1[j][r]) * va[j];
        }

#pragma unroll
    for (int m = 1; m < 16; m <<= 1)
#pragma unroll
        for (int r = 0; r < 8; ++r) {
            p0[r] += __shfl_xor(p0[r], m, 32);
            p1[r] += __shfl_xor(p1[r], m, 32);
        }

    if (lrow == 0) {
        const size_t mb0 = (size_t)b * T_ + twave + half * 8;       // tile 0 rows
        const size_t mb1 = (size_t)b * T_ + twave + 16 + half * 8;  // tile 1 rows
#pragma unroll
        for (int r = 0; r < 8; ++r) { e_out[mb0 + r] = p0[r]; e_out[mb1 + r] = p1[r]; }
    }
}

// ---------------------------------------------------------------------------
// Pass 2: per-batch max and sum(exp(e - max)) over T. One block per batch.
// ---------------------------------------------------------------------------
__global__ __launch_bounds__(256)
void softmax_reduce_kernel(const float* __restrict__ e, float* __restrict__ msum)
{
    __shared__ float red[256];
    const int b = blockIdx.x, tid = threadIdx.x;
    const float* eb = e + (size_t)b * T_;

    float m = -INFINITY;
    for (int t = tid; t < T_; t += 256) m = fmaxf(m, eb[t]);
    red[tid] = m;
    __syncthreads();
    for (int s = 128; s > 0; s >>= 1) {
        if (tid < s) red[tid] = fmaxf(red[tid], red[tid + s]);
        __syncthreads();
    }
    const float mb = red[0];
    __syncthreads();

    float s = 0.0f;
    for (int t = tid; t < T_; t += 256) s += expf(eb[t] - mb);
    red[tid] = s;
    __syncthreads();
    for (int k = 128; k > 0; k >>= 1) {
        if (tid < k) red[tid] += red[tid + k];
        __syncthreads();
    }
    if (tid == 0) { msum[2 * b] = mb; msum[2 * b + 1] = red[0]; }
}

// ---------------------------------------------------------------------------
// Pass 3: attention weights + coverage + per-chunk partial context sums.
// Grid: (T/512, B), block 256. Thread d accumulates channel d over the chunk.
// ---------------------------------------------------------------------------
__global__ __launch_bounds__(256)
void weights_ctx_kernel(const float* __restrict__ enc, const float* __restrict__ cov_in,
                        const float* __restrict__ e, const float* __restrict__ msum,
                        float* __restrict__ attn_out, float* __restrict__ cov_out,
                        float* __restrict__ ctx_part)
{
    __shared__ float wbuf[512];
    const int b = blockIdx.y, chunk = blockIdx.x, tid = threadIdx.x;
    const int t0 = chunk * 512;
    const float mb  = msum[2 * b];
    const float inv = 1.0f / msum[2 * b + 1];

    for (int i = tid; i < 512; i += 256) {
        const size_t gt = (size_t)b * T_ + t0 + i;
        const float w = expf(e[gt] - mb) * inv;
        attn_out[gt] = w;
        cov_out[gt]  = cov_in[gt] + w;
        wbuf[i] = w;
    }
    __syncthreads();

    const int d = tid;  // 256 threads <-> DE channels (coalesced)
    float acc = 0.0f;
    const float* ebase = enc + ((size_t)b * T_ + t0) * DE_ + d;
    for (int i = 0; i < 512; ++i) acc += wbuf[i] * ebase[(size_t)i * DE_];
    ctx_part[((size_t)b * 8 + chunk) * DE_ + d] = acc;
}

// ---------------------------------------------------------------------------
// Pass 4: fixed-order reduction of 8 chunk partials -> context [B, DE].
// ---------------------------------------------------------------------------
__global__ __launch_bounds__(256)
void ctx_final_kernel(const float* __restrict__ ctx_part, float* __restrict__ ctx_out)
{
    const int idx = blockIdx.x * 256 + threadIdx.x;
    if (idx >= B_ * DE_) return;
    const int b = idx / DE_, d = idx % DE_;
    float s = 0.0f;
#pragma unroll
    for (int c = 0; c < 8; ++c) s += ctx_part[((size_t)b * 8 + c) * DE_ + d];
    ctx_out[idx] = s;
}

extern "C" void kernel_launch(void* const* d_in, const int* in_sizes, int n_in,
                              void* d_out, int out_size, void* d_ws, size_t ws_size,
                              hipStream_t stream)
{
    (void)in_sizes; (void)n_in; (void)out_size; (void)ws_size;

    const float* enc = (const float*)d_in[0];   // [B,T,DE]
    const float* dec = (const float*)d_in[1];   // [B,T,DD]
    const float* cov = (const float*)d_in[2];   // [B,T,1]
    const float* W_a = (const float*)d_in[3];   // [DE,DD]
    const float* U_a = (const float*)d_in[4];   // [DD,DD]
    const float* v_a = (const float*)d_in[5];   // [DD,1]

    float* out      = (float*)d_out;
    float* ctx_out  = out;                        // [B,DE]   = 8192
    float* attn_out = out + B_ * DE_;             // [B,T,1]  = 131072
    float* cov_out  = attn_out + (size_t)B_ * T_; // [B,T,1]  = 131072

    float* ws     = (float*)d_ws;
    float* e_ws   = ws;                           // B*T = 131072 floats
    float* msum   = e_ws + (size_t)B_ * T_;       // 64 floats
    float* ctx_p  = msum + 64;                    // B*8*DE = 65536 floats

    dim3 g1(T_ / 256, B_);
    score_kernel<<<g1, 256, 0, stream>>>(enc, dec, W_a, U_a, v_a, e_ws);

    softmax_reduce_kernel<<<B_, 256, 0, stream>>>(e_ws, msum);

    dim3 g3(T_ / 512, B_);
    weights_ctx_kernel<<<g3, 256, 0, stream>>>(enc, cov, e_ws, msum,
                                               attn_out, cov_out, ctx_p);

    ctx_final_kernel<<<(B_ * DE_ + 255) / 256, 256, 0, stream>>>(ctx_p, ctx_out);
}
